// GATv2_65472481460436
// MI455X (gfx1250) — compile-verified, matrix operand order checked by reading, not run on hardware
//
#include <hip/hip_runtime.h>

typedef __attribute__((ext_vector_type(16))) __bf16       v16bf;
typedef __attribute__((ext_vector_type(8)))  float        v8f;
typedef __attribute__((ext_vector_type(8)))  unsigned int v8u;

#define D_IN  128
#define EMB   64
#define NH    8
#define HD    8
#define NC    128          // concatenated output cols: [Hs(64) | Hr(64)]

static __device__ __forceinline__ unsigned short f2bf(float f) {
    unsigned int u = __float_as_uint(f);
    unsigned int r = u + 0x7FFFu + ((u >> 16) & 1u);   // round-to-nearest-even
    return (unsigned short)(r >> 16);
}

// order-preserving float<->u32 encoding for atomic u32 max
static __device__ __forceinline__ unsigned int enc_f32(float f) {
    unsigned int u = __float_as_uint(f);
    return (u & 0x80000000u) ? ~u : (u | 0x80000000u);
}
static __device__ __forceinline__ float dec_f32(unsigned int e) {
    return (e & 0x80000000u) ? __uint_as_float(e & 0x7FFFFFFFu)
                             : __uint_as_float(~e);
}

static __device__ __forceinline__ float mish(float x) {
    float sp = (x > 20.0f) ? x : log1pf(__expf(x));
    return x * tanhf(sp);
}

// ---------------------------------------------------------------------------
// K0: zero output / denom / segmax
// ---------------------------------------------------------------------------
__global__ __launch_bounds__(256) void k_init(float* __restrict__ out, int n_out,
                                              float* __restrict__ denom,
                                              unsigned int* __restrict__ segmax,
                                              int n_seg) {
    int t = blockIdx.x * 256 + threadIdx.x;
    if (t < n_out) out[t] = 0.0f;
    if (t < n_seg) { denom[t] = 0.0f; segmax[t] = 0u; }   // 0 == -inf in encoding
}

// ---------------------------------------------------------------------------
// K1: node projection  Hsr[n, 0:64]=nodes@Ws+bs, Hsr[n,64:128]=nodes@Wr+br
//     bf16 WMMA, fp32 accumulate. 64 rows per 256-thread workgroup.
//
// LDS is pre-swizzled into WMMA fragment-native layout:
//   frag[fragId][h2][lane][4 dwords]  (two 16B-aligned b128 chunks per lane)
// so the inner loop is pure contiguous ds_load_b128 + v_wmma, no address ALU.
//
// ISA 16-bit A/B fragment layout (lane = half*16+idx):
//   pack p (0..7) holds K = (p<4?0:16) + half*8 + (p&3)*2 and +1.
//   For element K: hi16=K>>4 (==p>>2), half=(K&15)>>3, pd=(K&7)>>1, slot=K&1.
// ---------------------------------------------------------------------------
__global__ __launch_bounds__(256) void k_node_proj(
    const float* __restrict__ nodes,
    const float* __restrict__ Ws, const float* __restrict__ bs,
    const float* __restrict__ Wr, const float* __restrict__ br,
    float* __restrict__ Hsr, int n_nodes)
{
    // B fragments: fragId = kk4*8 + colblock(wave)  -> 32 KB
    __shared__ __align__(16) unsigned int Bfrag[32][2][32][4];
    // A fragments: fragId = kk4*4 + rowblock        -> 16 KB
    __shared__ __align__(16) unsigned int Afrag[16][2][32][4];

    const int t    = threadIdx.x;
    const int row0 = blockIdx.x * 64;

    // ---- stage combined weights [k=128][n=128], packing bf16 pairs ----
    // 128 cols x 64 k-pairs = 8192 dwords, 32 per thread
    for (int i = 0; i < 32; ++i) {
        int flat = i * 256 + t;
        int n  = flat >> 6;            // 0..127
        int k2 = flat & 63;            // k-pair
        int k  = k2 * 2;
        float v0, v1;
        if (n < EMB) { v0 = Ws[k * EMB + n];         v1 = Ws[(k + 1) * EMB + n]; }
        else         { v0 = Wr[k * EMB + (n - EMB)]; v1 = Wr[(k + 1) * EMB + (n - EMB)]; }
        unsigned int pack = (unsigned int)f2bf(v0) | ((unsigned int)f2bf(v1) << 16);
        int kk4  = k >> 5;
        int ko   = k & 31;
        int hi16 = ko >> 4;
        int rem  = ko & 15;
        int half = rem >> 3;
        int pd   = (rem & 7) >> 1;
        Bfrag[kk4 * 8 + (n >> 4)][hi16][half * 16 + (n & 15)][pd] = pack;
    }
    // ---- stage A tile: 64 rows x 64 k-pairs = 4096 dwords, 16 per thread ----
    for (int i = 0; i < 16; ++i) {
        int flat = i * 256 + t;
        int m  = flat >> 6;            // 0..63
        int k2 = flat & 63;
        int k  = k2 * 2;
        int gr = row0 + m;
        float v0 = 0.0f, v1 = 0.0f;
        if (gr < n_nodes) {
            float2 vv = ((const float2*)(nodes + (size_t)gr * D_IN))[k2];
            v0 = vv.x; v1 = vv.y;
        }
        unsigned int pack = (unsigned int)f2bf(v0) | ((unsigned int)f2bf(v1) << 16);
        int kk4  = k >> 5;
        int ko   = k & 31;
        int hi16 = ko >> 4;
        int rem  = ko & 15;
        int half = rem >> 3;
        int pd   = (rem & 7) >> 1;
        Afrag[kk4 * 4 + (m >> 4)][hi16][half * 16 + (m & 15)][pd] = pack;
    }
    __syncthreads();

    const int lane = t & 31;                     // wave32
    const int wave = t >> 5;                     // 8 waves: one 16-col block each
    const int half = lane >> 4;
    const int idx  = lane & 15;
    const int col  = wave * 16 + idx;

    float bias = (col < EMB) ? bs[col] : br[col - EMB];
    v8f c[4];
    #pragma unroll
    for (int rb = 0; rb < 4; ++rb)
        #pragma unroll
        for (int v = 0; v < 8; ++v) c[rb][v] = bias;

    #pragma unroll
    for (int kk4 = 0; kk4 < 4; ++kk4) {
        v8u bu;
        #pragma unroll
        for (int j = 0; j < 4; ++j) bu[j]     = Bfrag[kk4 * 8 + wave][0][lane][j];
        #pragma unroll
        for (int j = 0; j < 4; ++j) bu[4 + j] = Bfrag[kk4 * 8 + wave][1][lane][j];
        v16bf b = __builtin_bit_cast(v16bf, bu);
        // reuse B registers across the 4 row-blocks
        #pragma unroll
        for (int rb = 0; rb < 4; ++rb) {
            v8u au;
            #pragma unroll
            for (int j = 0; j < 4; ++j) au[j]     = Afrag[kk4 * 4 + rb][0][lane][j];
            #pragma unroll
            for (int j = 0; j < 4; ++j) au[4 + j] = Afrag[kk4 * 4 + rb][1][lane][j];
            v16bf a = __builtin_bit_cast(v16bf, au);
            c[rb] = __builtin_amdgcn_wmma_f32_16x16x32_bf16(
                        false, a, false, b, (short)0, c[rb], false, false);
        }
    }

    #pragma unroll
    for (int rb = 0; rb < 4; ++rb)
        #pragma unroll
        for (int v = 0; v < 8; ++v) {
            int gr = row0 + rb * 16 + v + 8 * half;   // C/D row layout
            if (gr < n_nodes) Hsr[(size_t)gr * NC + col] = c[rb][v];
        }
}

// ---------------------------------------------------------------------------
// K2: per-(edge,head) attention logit + atomic segment max (encoded u32)
// ---------------------------------------------------------------------------
__global__ __launch_bounds__(256) void k_edge_max(
    const float* __restrict__ Hsr,
    const int* __restrict__ snd, const int* __restrict__ rcv,
    const float* __restrict__ wa, const float* __restrict__ ba,
    unsigned int* __restrict__ segmax, long long n_eh)
{
    long long t = (long long)blockIdx.x * 256 + threadIdx.x;
    if (t >= n_eh) return;
    int e = (int)(t >> 3), h = (int)(t & 7);
    int s = snd[e], r = rcv[e];
    const float* hs = Hsr + (size_t)s * NC + h * HD;
    const float* hr = Hsr + (size_t)r * NC + EMB + h * HD;
    float logit = ba[0];
    #pragma unroll
    for (int d = 0; d < HD; ++d)
        logit += wa[d] * mish(hs[d] + hr[d]);
    atomicMax(&segmax[r * NH + h], enc_f32(logit));
}

// ---------------------------------------------------------------------------
// K3: recompute logit, u = exp(logit - max); atomic add denom and
//     unnormalized weighted sender features into out (normalized later).
// ---------------------------------------------------------------------------
__global__ __launch_bounds__(256) void k_edge_scatter(
    const float* __restrict__ Hsr,
    const int* __restrict__ snd, const int* __restrict__ rcv,
    const float* __restrict__ wa, const float* __restrict__ ba,
    const unsigned int* __restrict__ segmax,
    float* __restrict__ denom, float* __restrict__ out, long long n_eh)
{
    long long t = (long long)blockIdx.x * 256 + threadIdx.x;
    if (t >= n_eh) return;
    int e = (int)(t >> 3), h = (int)(t & 7);
    int s = snd[e], r = rcv[e];
    const float* hs = Hsr + (size_t)s * NC + h * HD;
    const float* hr = Hsr + (size_t)r * NC + EMB + h * HD;
    float logit = ba[0];
    #pragma unroll
    for (int d = 0; d < HD; ++d)
        logit += wa[d] * mish(hs[d] + hr[d]);
    float mx = dec_f32(segmax[r * NH + h]);
    float u  = __expf(logit - mx);
    atomicAdd(&denom[r * NH + h], u);
    float* o = out + (size_t)r * EMB + h * HD;
    #pragma unroll
    for (int d = 0; d < HD; ++d)
        atomicAdd(&o[d], u * hs[d]);
}

// ---------------------------------------------------------------------------
// K4: normalize aggregated messages by softmax denominator
// ---------------------------------------------------------------------------
__global__ __launch_bounds__(256) void k_normalize(
    float* __restrict__ out, const float* __restrict__ denom, int n_total)
{
    int t = blockIdx.x * 256 + threadIdx.x;
    if (t >= n_total) return;
    int n = t >> 6;              // node
    int h = (t >> 3) & 7;        // head
    float d = denom[n * NH + h];
    out[t] = (d > 0.0f) ? out[t] / d : 0.0f;
}

// ---------------------------------------------------------------------------
extern "C" void kernel_launch(void* const* d_in, const int* in_sizes, int n_in,
                              void* d_out, int out_size, void* d_ws, size_t ws_size,
                              hipStream_t stream) {
    const float* nodes = (const float*)d_in[0];
    const int*   snd   = (const int*)d_in[1];
    const int*   rcv   = (const int*)d_in[2];
    const float* Ws    = (const float*)d_in[3];
    const float* bs    = (const float*)d_in[4];
    const float* Wr    = (const float*)d_in[5];
    const float* br    = (const float*)d_in[6];
    const float* wa    = (const float*)d_in[7];
    const float* ba    = (const float*)d_in[8];
    float*       out   = (float*)d_out;

    const int n_nodes = in_sizes[0] / D_IN;
    const int n_edges = in_sizes[1];

    // workspace layout
    char* ws = (char*)d_ws;
    float*        Hsr    = (float*)ws;                                   // n_nodes*128 f32
    unsigned int* segmax = (unsigned int*)(ws + (size_t)n_nodes * NC * 4);
    float*        denom  = (float*)(ws + (size_t)n_nodes * NC * 4
                                       + (size_t)n_nodes * NH * 4);

    const int n_out = n_nodes * EMB;
    const int n_seg = n_nodes * NH;
    const long long n_eh = (long long)n_edges * NH;

    k_init<<<(n_out + 255) / 256, 256, 0, stream>>>(out, n_out, denom, segmax, n_seg);

    k_node_proj<<<(n_nodes + 63) / 64, 256, 0, stream>>>(
        nodes, Ws, bs, Wr, br, Hsr, n_nodes);

    int eh_blocks = (int)((n_eh + 255) / 256);
    k_edge_max<<<eh_blocks, 256, 0, stream>>>(Hsr, snd, rcv, wa, ba, segmax, n_eh);

    k_edge_scatter<<<eh_blocks, 256, 0, stream>>>(Hsr, snd, rcv, wa, ba,
                                                  segmax, denom, out, n_eh);

    k_normalize<<<(n_out + 255) / 256, 256, 0, stream>>>(out, denom, n_out);
}
